// MultiHeadAttention_77781857730869
// MI455X (gfx1250) — compile-verified, matrix operand order checked by reading, not run on hardware
//
#include <hip/hip_runtime.h>
#include <hip/hip_bf16.h>

// ---------------------------------------------------------------------------
// MultiHeadAttention forward for MI455X (gfx1250, wave32, WMMA).
// bf16 WMMA (v_wmma_f32_16x16x32_bf16) for all GEMMs, f32 accumulation.
// Double-buffered LDS staging; async global->LDS (ASYNCcnt) when the
// toolchain declares the gfx1250 builtins, sync copy otherwise.
// attn_w fp32 lives in d_out and is passed over 3x (write, softmax rw, read)
// = ~1.6GB HBM traffic which dominates the roofline (~70us @ 23.3 TB/s).
// ---------------------------------------------------------------------------

#define EMB    768
#define HEADS  12
#define DEPTH  64
#define BATCH  8
#define SEQ    1024
#define MROWS  (BATCH * SEQ)          // 8192
#define SCALE  0.03608439182435161f   // 1/sqrt(768)

typedef __bf16 bf16_t;
typedef __attribute__((ext_vector_type(16))) __bf16 v16bf;
typedef __attribute__((ext_vector_type(8)))  __bf16 v8bf;
typedef __attribute__((ext_vector_type(8)))  float  v8f;
typedef __attribute__((ext_vector_type(4)))  int    v4i;

#define BT_M  128
#define KSTEP 32

#define AS1 __attribute__((address_space(1)))
#define AS3 __attribute__((address_space(3)))

#if defined(__gfx1250__) && \
    __has_builtin(__builtin_amdgcn_global_load_async_to_lds_b128) && \
    __has_builtin(__builtin_amdgcn_s_wait_asynccnt)
#define USE_ASYNC 1
#else
#define USE_ASYNC 0
#endif

__device__ inline void async_wait() {
#if USE_ASYNC
  __builtin_amdgcn_s_wait_asynccnt(0);
#endif
}

// ---------------------------------------------------------------------------
// LDS staging helpers (256 threads cooperate, tiles are [rows x 32] bf16).
// ---------------------------------------------------------------------------

// 128x32 bf16 tile: each thread owns 16 elements (32B).
__device__ inline void stage_bf16_128x32(bf16_t* s, const bf16_t* __restrict__ g, int ldg) {
  int t   = threadIdx.x;
  int row = t >> 1;
  int kb  = (t & 1) << 4;
  const bf16_t* src = g + (size_t)row * ldg + kb;
  bf16_t* dst = s + row * KSTEP + kb;
#if USE_ASYNC
  __builtin_amdgcn_global_load_async_to_lds_b128((AS1 v4i*)src, (AS3 v4i*)dst, 0, 0);
  __builtin_amdgcn_global_load_async_to_lds_b128((AS1 v4i*)src, (AS3 v4i*)dst, 16, 0);
#else
  const uint4* s4 = (const uint4*)src;
  uint4* d4 = (uint4*)dst;
  d4[0] = s4[0];
  d4[1] = s4[1];
#endif
}

// 64x32 bf16 tile: each thread owns 8 elements (16B).
__device__ inline void stage_bf16_64x32(bf16_t* s, const bf16_t* __restrict__ g, int ldg) {
  int t   = threadIdx.x;
  int row = t >> 2;
  int kb  = (t & 3) << 3;
  const bf16_t* src = g + (size_t)row * ldg + kb;
  bf16_t* dst = s + row * KSTEP + kb;
#if USE_ASYNC
  __builtin_amdgcn_global_load_async_to_lds_b128((AS1 v4i*)src, (AS3 v4i*)dst, 0, 0);
#else
  *(uint4*)dst = *(const uint4*)src;
#endif
}

// 128x32 f32 -> bf16 tile (conversion in flight; stays on the VALU path).
__device__ inline void stage_f32_128x32(bf16_t* s, const float* __restrict__ g, int ldg) {
  int t   = threadIdx.x;
  int row = t >> 1;
  int kb  = (t & 1) << 4;
  const float* src = g + (size_t)row * ldg + kb;
  __builtin_prefetch((const char*)src + KSTEP * sizeof(float), 0, 0);
  float4 f[4];
#pragma unroll
  for (int i = 0; i < 4; ++i) f[i] = ((const float4*)src)[i];
  union { bf16_t h[16]; uint4 v[2]; } u;
#pragma unroll
  for (int i = 0; i < 4; ++i) {
    u.h[4 * i + 0] = (bf16_t)f[i].x;
    u.h[4 * i + 1] = (bf16_t)f[i].y;
    u.h[4 * i + 2] = (bf16_t)f[i].z;
    u.h[4 * i + 3] = (bf16_t)f[i].w;
  }
  uint4* dst = (uint4*)(s + row * KSTEP + kb);
  dst[0] = u.v[0];
  dst[1] = u.v[1];
}

// ---------------------------------------------------------------------------
// Fragment load from row-major [rows x 32] bf16 LDS tile, per ISA A/B layout:
// lanes 0-15 (row m): K 0..7, 16..23; lanes 16-31 (row m): K 8..15, 24..31.
// ---------------------------------------------------------------------------
__device__ inline v16bf frag_ld(const bf16_t* base) {
  int lane = threadIdx.x & 31;
  const bf16_t* p = base + (size_t)(lane & 15) * KSTEP + (((lane >> 4) & 1) << 3);
  union { v8bf v[2]; v16bf f; } u;
  u.v[0] = *(const v8bf*)(p);
  u.v[1] = *(const v8bf*)(p + 16);
  return u.f;
}

// One K-step of a 32x(CT*16) wave tile: 2 x CT WMMAs 16x16x32 bf16->f32.
template <int CT>
__device__ inline void mma_step(const bf16_t* sA, const bf16_t* sB,
                                int wm, int wn, v8f acc[2][CT]) {
  v16bf a[2], b[CT];
  a[0] = frag_ld(sA + (wm * 32 +  0) * KSTEP);
  a[1] = frag_ld(sA + (wm * 32 + 16) * KSTEP);
#pragma unroll
  for (int j = 0; j < CT; ++j)
    b[j] = frag_ld(sB + (wn * CT + j) * 16 * KSTEP);
#pragma unroll
  for (int i = 0; i < 2; ++i)
#pragma unroll
    for (int j = 0; j < CT; ++j)
      acc[i][j] = __builtin_amdgcn_wmma_f32_16x16x32_bf16(
          false, a[i], false, b[j], (short)0, acc[i][j], false, false);
}

template <int CT>
__device__ inline void acc_zero(v8f acc[2][CT]) {
#pragma unroll
  for (int i = 0; i < 2; ++i)
#pragma unroll
    for (int j = 0; j < CT; ++j)
#pragma unroll
      for (int e = 0; e < 8; ++e) acc[i][j][e] = 0.0f;
}

// ---------------------------------------------------------------------------
// Kernel: fp32 -> bf16 weight conversion (4 elems/thread).
// ---------------------------------------------------------------------------
__global__ __launch_bounds__(256) void k_cvt_w(const float* __restrict__ w,
                                               bf16_t* __restrict__ o, int n4) {
  int i = blockIdx.x * blockDim.x + threadIdx.x;
  if (i < n4) {
    float4 f = ((const float4*)w)[i];
    union { bf16_t h[4]; uint2 v; } u;
    u.h[0] = (bf16_t)f.x; u.h[1] = (bf16_t)f.y;
    u.h[2] = (bf16_t)f.z; u.h[3] = (bf16_t)f.w;
    ((uint2*)o)[i] = u.v;
  }
}

// ---------------------------------------------------------------------------
// Kernel: projection  y = x @ W^T + b, head-split bf16 output.
// VMODE==0: out[b,h,l,d] (q/k);  VMODE==1: out[b,h,d,l] (V transposed).
// Block tile 128x128, waves 4x2, wave tile 32x64.  grid = (EMB/128, MROWS/128)
// ---------------------------------------------------------------------------
template <int VMODE>
__global__ __launch_bounds__(256)
void k_proj(const float* __restrict__ X, const bf16_t* __restrict__ Wbf,
            const float* __restrict__ bias, bf16_t* __restrict__ out) {
  constexpr int BN = 128, CT = BN / 32;
  __shared__ __align__(16) bf16_t sA[2][BT_M * KSTEP];
  __shared__ __align__(16) bf16_t sB[2][BN * KSTEP];
  int m0 = blockIdx.y * BT_M;
  int n0 = blockIdx.x * BN;
  int wave = threadIdx.x >> 5, wm = wave & 3, wn = wave >> 2;
  v8f acc[2][CT];
  acc_zero<CT>(acc);
  const float*  Abase = X + (size_t)m0 * EMB;
  const bf16_t* Bbase = Wbf + (size_t)n0 * EMB;
  constexpr int NK = EMB / KSTEP;
  stage_f32_128x32(sA[0], Abase, EMB);
  stage_bf16_128x32(sB[0], Bbase, EMB);
  async_wait();
  __syncthreads();
  for (int i = 0; i < NK; ++i) {
    int cur = i & 1, nxt = cur ^ 1;
    if (i + 1 < NK) {
      stage_f32_128x32(sA[nxt], Abase + (i + 1) * KSTEP, EMB);
      stage_bf16_128x32(sB[nxt], Bbase + (i + 1) * KSTEP, EMB);
    }
    mma_step<CT>(sA[cur], sB[cur], wm, wn, acc);
    async_wait();
    __syncthreads();
  }
  int lane = threadIdx.x & 31;
  int cn = lane & 15, cmb = ((lane >> 4) & 1) << 3;
#pragma unroll
  for (int ti = 0; ti < 2; ++ti)
#pragma unroll
    for (int tj = 0; tj < CT; ++tj) {
      int gn = n0 + (wn * CT + tj) * 16 + cn;
      float bv = bias[gn];
      int h = gn >> 6, d = gn & 63;
#pragma unroll
      for (int r = 0; r < 8; ++r) {
        int gm = m0 + wm * 32 + ti * 16 + cmb + r;
        int b = gm >> 10, l = gm & 1023;
        float val = acc[ti][tj][r] + bv;
        size_t off = VMODE
            ? (((size_t)(b * HEADS + h) * DEPTH + d) * SEQ + l)
            : (((size_t)(b * HEADS + h) * SEQ + l) * DEPTH + d);
        out[off] = (bf16_t)val;
      }
    }
}

// ---------------------------------------------------------------------------
// Kernel: scores S = (Q K^T) * scale -> fp32 attn_w region of d_out.
// Block tile 128x128.  grid = (SEQ/128, SEQ/128, BATCH*HEADS)
// ---------------------------------------------------------------------------
__global__ __launch_bounds__(256)
void k_score(const bf16_t* __restrict__ qh, const bf16_t* __restrict__ kh,
             float* __restrict__ attw) {
  constexpr int BN = 128, CT = BN / 32;
  __shared__ __align__(16) bf16_t sA[2][BT_M * KSTEP];
  __shared__ __align__(16) bf16_t sB[2][BN * KSTEP];
  int bh = blockIdx.z;
  int m0 = blockIdx.y * BT_M;
  int n0 = blockIdx.x * BN;
  const bf16_t* Abase = qh + (size_t)bh * SEQ * DEPTH + (size_t)m0 * DEPTH;
  const bf16_t* Bbase = kh + (size_t)bh * SEQ * DEPTH + (size_t)n0 * DEPTH;
  float* O = attw + (size_t)bh * SEQ * SEQ;
  int wave = threadIdx.x >> 5, wm = wave & 3, wn = wave >> 2;
  v8f acc[2][CT];
  acc_zero<CT>(acc);
  constexpr int NK = DEPTH / KSTEP;   // 2
  stage_bf16_128x32(sA[0], Abase, DEPTH);
  stage_bf16_128x32(sB[0], Bbase, DEPTH);
  async_wait();
  __syncthreads();
#pragma unroll
  for (int i = 0; i < NK; ++i) {
    int cur = i & 1, nxt = cur ^ 1;
    if (i + 1 < NK) {
      stage_bf16_128x32(sA[nxt], Abase + (i + 1) * KSTEP, DEPTH);
      stage_bf16_128x32(sB[nxt], Bbase + (i + 1) * KSTEP, DEPTH);
    }
    mma_step<CT>(sA[cur], sB[cur], wm, wn, acc);
    async_wait();
    __syncthreads();
  }
  int lane = threadIdx.x & 31;
  int cn = lane & 15, cmb = ((lane >> 4) & 1) << 3;
#pragma unroll
  for (int ti = 0; ti < 2; ++ti)
#pragma unroll
    for (int tj = 0; tj < CT; ++tj) {
      int gn = n0 + (wn * CT + tj) * 16 + cn;
#pragma unroll
      for (int r = 0; r < 8; ++r) {
        int gm = m0 + wm * 32 + ti * 16 + cmb + r;
        O[(size_t)gm * SEQ + gn] = acc[ti][tj][r] * SCALE;
      }
    }
}

// ---------------------------------------------------------------------------
// Kernel: in-place row softmax over SEQ=1024 (one block per row).
// ---------------------------------------------------------------------------
__global__ __launch_bounds__(256) void k_softmax(float* __restrict__ attw) {
  __shared__ float red[256];
  float* p = attw + (size_t)blockIdx.x * SEQ;
  int t = threadIdx.x;
  float4 x = ((float4*)p)[t];
  float mx = fmaxf(fmaxf(x.x, x.y), fmaxf(x.z, x.w));
  red[t] = mx;
  __syncthreads();
  for (int s = 128; s > 0; s >>= 1) {
    if (t < s) red[t] = fmaxf(red[t], red[t + s]);
    __syncthreads();
  }
  mx = red[0];
  __syncthreads();
  float4 e;
  e.x = __expf(x.x - mx); e.y = __expf(x.y - mx);
  e.z = __expf(x.z - mx); e.w = __expf(x.w - mx);
  red[t] = e.x + e.y + e.z + e.w;
  __syncthreads();
  for (int s = 128; s > 0; s >>= 1) {
    if (t < s) red[t] += red[t + s];
    __syncthreads();
  }
  float inv = 1.0f / red[0];
  e.x *= inv; e.y *= inv; e.z *= inv; e.w *= inv;
  ((float4*)p)[t] = e;
}

// ---------------------------------------------------------------------------
// Kernel: context O = P @ V.  A = fp32 probs (bf16-converted in staging),
// B = vT (d-major).  Block tile 128x64.  grid = (1, SEQ/128, BATCH*HEADS)
// ---------------------------------------------------------------------------
__global__ __launch_bounds__(256)
void k_context(const float* __restrict__ attw, const bf16_t* __restrict__ vt,
               bf16_t* __restrict__ ctx) {
  constexpr int BN = 64, CT = BN / 32;
  __shared__ __align__(16) bf16_t sA[2][BT_M * KSTEP];
  __shared__ __align__(16) bf16_t sB[2][BN * KSTEP];
  int bh = blockIdx.z;
  int b = bh / HEADS, h = bh % HEADS;
  int m0 = blockIdx.y * BT_M;
  const float*  Abase = attw + (size_t)bh * SEQ * SEQ + (size_t)m0 * SEQ;
  const bf16_t* Bbase = vt + (size_t)bh * DEPTH * SEQ;
  int wave = threadIdx.x >> 5, wm = wave & 3, wn = wave >> 2;
  v8f acc[2][CT];
  acc_zero<CT>(acc);
  constexpr int NK = SEQ / KSTEP;   // 32
  stage_f32_128x32(sA[0], Abase, SEQ);
  stage_bf16_64x32(sB[0], Bbase, SEQ);
  async_wait();
  __syncthreads();
  for (int i = 0; i < NK; ++i) {
    int cur = i & 1, nxt = cur ^ 1;
    if (i + 1 < NK) {
      stage_f32_128x32(sA[nxt], Abase + (i + 1) * KSTEP, SEQ);
      stage_bf16_64x32(sB[nxt], Bbase + (i + 1) * KSTEP, SEQ);
    }
    mma_step<CT>(sA[cur], sB[cur], wm, wn, acc);
    async_wait();
    __syncthreads();
  }
  int lane = threadIdx.x & 31;
  int cn = lane & 15, cmb = ((lane >> 4) & 1) << 3;
#pragma unroll
  for (int ti = 0; ti < 2; ++ti)
#pragma unroll
    for (int tj = 0; tj < CT; ++tj) {
      int gn = (wn * CT + tj) * 16 + cn;   // d in 0..63
#pragma unroll
      for (int r = 0; r < 8; ++r) {
        int gm = m0 + wm * 32 + ti * 16 + cmb + r;   // l
        ctx[(size_t)(b * SEQ + gm) * EMB + h * DEPTH + gn] = (bf16_t)acc[ti][tj][r];
      }
    }
}

// ---------------------------------------------------------------------------
// Kernel: dense  y = ctx @ Wd^T + bd -> fp32 attention output.
// Block tile 128x128.  grid = (EMB/128, MROWS/128)
// ---------------------------------------------------------------------------
__global__ __launch_bounds__(256)
void k_dense(const bf16_t* __restrict__ ctx, const bf16_t* __restrict__ wd,
             const float* __restrict__ bias, float* __restrict__ out) {
  constexpr int BN = 128, CT = BN / 32;
  __shared__ __align__(16) bf16_t sA[2][BT_M * KSTEP];
  __shared__ __align__(16) bf16_t sB[2][BN * KSTEP];
  int m0 = blockIdx.y * BT_M;
  int n0 = blockIdx.x * BN;
  const bf16_t* Abase = ctx + (size_t)m0 * EMB;
  const bf16_t* Bbase = wd + (size_t)n0 * EMB;
  int wave = threadIdx.x >> 5, wm = wave & 3, wn = wave >> 2;
  v8f acc[2][CT];
  acc_zero<CT>(acc);
  constexpr int NK = EMB / KSTEP;
  stage_bf16_128x32(sA[0], Abase, EMB);
  stage_bf16_128x32(sB[0], Bbase, EMB);
  async_wait();
  __syncthreads();
  for (int i = 0; i < NK; ++i) {
    int cur = i & 1, nxt = cur ^ 1;
    if (i + 1 < NK) {
      stage_bf16_128x32(sA[nxt], Abase + (i + 1) * KSTEP, EMB);
      stage_bf16_128x32(sB[nxt], Bbase + (i + 1) * KSTEP, EMB);
    }
    mma_step<CT>(sA[cur], sB[cur], wm, wn, acc);
    async_wait();
    __syncthreads();
  }
  int lane = threadIdx.x & 31;
  int cn = lane & 15, cmb = ((lane >> 4) & 1) << 3;
#pragma unroll
  for (int ti = 0; ti < 2; ++ti)
#pragma unroll
    for (int tj = 0; tj < CT; ++tj) {
      int gn = n0 + (wn * CT + tj) * 16 + cn;
      float bv = bias[gn];
#pragma unroll
      for (int r = 0; r < 8; ++r) {
        int gm = m0 + wm * 32 + ti * 16 + cmb + r;
        out[(size_t)gm * EMB + gn] = acc[ti][tj][r] + bv;
      }
    }
}

// ---------------------------------------------------------------------------
// Launch.  Workspace (~52.5 MB):
//   4x bf16 weights (4*768*768*2) | qh | kh | vT | ctx  (bf16 each)
// ---------------------------------------------------------------------------
extern "C" void kernel_launch(void* const* d_in, const int* in_sizes, int n_in,
                              void* d_out, int out_size, void* d_ws, size_t ws_size,
                              hipStream_t stream) {
  (void)in_sizes; (void)n_in; (void)out_size; (void)ws_size;
  const float* v_in = (const float*)d_in[0];
  const float* k_in = (const float*)d_in[1];
  const float* q_in = (const float*)d_in[2];
  const float* wq_w = (const float*)d_in[3];
  const float* wq_b = (const float*)d_in[4];
  const float* wk_w = (const float*)d_in[5];
  const float* wk_b = (const float*)d_in[6];
  const float* wv_w = (const float*)d_in[7];
  const float* wv_b = (const float*)d_in[8];
  const float* wd_w = (const float*)d_in[9];
  const float* wd_b = (const float*)d_in[10];

  const size_t W_ELEMS  = (size_t)EMB * EMB;
  const size_t QH_ELEMS = (size_t)BATCH * HEADS * SEQ * DEPTH;

  char* ws = (char*)d_ws;
  bf16_t* wq_bf = (bf16_t*)ws;
  bf16_t* wk_bf = wq_bf + W_ELEMS;
  bf16_t* wv_bf = wk_bf + W_ELEMS;
  bf16_t* wd_bf = wv_bf + W_ELEMS;
  bf16_t* qh    = wd_bf + W_ELEMS;
  bf16_t* kh    = qh + QH_ELEMS;
  bf16_t* vt    = kh + QH_ELEMS;
  bf16_t* ctx   = vt + QH_ELEMS;

  float* out_att = (float*)d_out;
  float* attw    = out_att + (size_t)MROWS * EMB;

  int n4 = (int)(W_ELEMS / 4);
  int cvb = (n4 + 255) / 256;
  k_cvt_w<<<cvb, 256, 0, stream>>>(wq_w, wq_bf, n4);
  k_cvt_w<<<cvb, 256, 0, stream>>>(wk_w, wk_bf, n4);
  k_cvt_w<<<cvb, 256, 0, stream>>>(wv_w, wv_bf, n4);
  k_cvt_w<<<cvb, 256, 0, stream>>>(wd_w, wd_bf, n4);

  dim3 gProj(EMB / 128, MROWS / BT_M);
  k_proj<0><<<gProj, 256, 0, stream>>>(q_in, wq_bf, wq_b, qh);
  k_proj<0><<<gProj, 256, 0, stream>>>(k_in, wk_bf, wk_b, kh);
  k_proj<1><<<gProj, 256, 0, stream>>>(v_in, wv_bf, wv_b, vt);

  dim3 gScore(SEQ / 128, SEQ / BT_M, BATCH * HEADS);
  k_score<<<gScore, 256, 0, stream>>>(qh, kh, attw);

  k_softmax<<<BATCH * HEADS * SEQ, 256, 0, stream>>>(attw);

  dim3 gCtx(1, SEQ / BT_M, BATCH * HEADS);
  k_context<<<gCtx, 256, 0, stream>>>(attw, vt, ctx);

  k_dense<<<gProj, 256, 0, stream>>>(ctx, wd_bf, wd_b, out_att);
}
